// CPQuadRankLayer_6253472383189
// MI455X (gfx1250) — matrix operand order repaired and checked by reading, not compiled, not run on hardware
//
#include <hip/hip_runtime.h>

#define B_    32
#define N_    1024
#define DIM_  256
#define RANK_ 32
#define LN_EPS 1e-5f

typedef __attribute__((ext_vector_type(16))) __bf16 v16bf;
typedef __attribute__((ext_vector_type(8)))  __bf16 v8bf;
typedef __attribute__((ext_vector_type(8)))  float  v8f;

// ---------------- LDS layout (bytes) ----------------
// xn   : ushort[128][264]  (4 children * 32 batch rows, padded stride)
// fc   : ushort[4][32][264] (rank-major factor matrices, bf16)
// fout : ushort[256][40]   (factor_out transposed to [o][r], padded)
// mg   : ushort[32][40]    (merged, A-fragment layout for GEMM2)
// res  : float[32][260]    (residual, padded)
// pm   : float[4][32][33]  (P_c projections)
// sc   : float[32]         (scale)
#define XN_OFF    0
#define FC_OFF    67584
#define FOUT_OFF  135168
#define MG_OFF    155648
#define RES_OFF   158208
#define P_OFF     191488
#define SC_OFF    208384
#define LDS_BYTES 208512

static __device__ __forceinline__ unsigned short f2bf(float f) {
  unsigned int u = __float_as_uint(f);
  u += 0x7FFFu + ((u >> 16) & 1u);          // round-to-nearest-even
  return (unsigned short)(u >> 16);
}
static __device__ __forceinline__ unsigned int pk2(float a, float b) {
  return (unsigned int)f2bf(a) | ((unsigned int)f2bf(b) << 16);
}

// A-matrix 16x32 bf16 fragment: per-lane 16 consecutive K values of one row
static __device__ __forceinline__ v16bf frag_a(const unsigned short* p) {
  v8bf lo = *(const v8bf*)p;
  v8bf hi = *(const v8bf*)(p + 8);
  return __builtin_shufflevector(lo, hi, 0,1,2,3,4,5,6,7,8,9,10,11,12,13,14,15);
}
// B-matrix 32x16 bf16 fragment: per-lane 16 consecutive K values of one column
static __device__ __forceinline__ v16bf frag_b(const unsigned short* p) {
  v8bf lo = *(const v8bf*)p;
  v8bf hi = *(const v8bf*)(p + 8);
  return __builtin_shufflevector(lo, hi, 0,1,2,3,4,5,6,7,8,9,10,11,12,13,14,15);
}

extern "C" __global__ __launch_bounds__(256)
void cp_quad_rank_wmma(const float* __restrict__ x,
                       const float* __restrict__ f_tl, const float* __restrict__ f_tr,
                       const float* __restrict__ f_bl, const float* __restrict__ f_br,
                       const float* __restrict__ f_out,
                       const float* __restrict__ scale,
                       const float* __restrict__ gamma, const float* __restrict__ beta,
                       float* __restrict__ out)
{
  extern __shared__ unsigned char smem[];
  unsigned short* xn   = (unsigned short*)(smem + XN_OFF);
  unsigned short* fc   = (unsigned short*)(smem + FC_OFF);
  unsigned short* fout = (unsigned short*)(smem + FOUT_OFF);
  unsigned short* mg   = (unsigned short*)(smem + MG_OFF);
  float* res = (float*)(smem + RES_OFF);
  float* pm  = (float*)(smem + P_OFF);
  float* sc  = (float*)(smem + SC_OFF);

  const int n    = blockIdx.x;
  const int tid  = threadIdx.x;
  const int wave = tid >> 5;
  const int lane = tid & 31;
  const int l15  = lane & 15;
  const bool hiL = lane >= 16;

  const float* fsrc0 = f_tl  + (size_t)n * RANK_ * DIM_;
  const float* fsrc1 = f_tr  + (size_t)n * RANK_ * DIM_;
  const float* fsrc2 = f_bl  + (size_t)n * RANK_ * DIM_;
  const float* fsrc3 = f_br  + (size_t)n * RANK_ * DIM_;
  const float* fsrco = f_out + (size_t)n * RANK_ * DIM_;
  __builtin_prefetch(fsrco, 0, 1);
  __builtin_prefetch(fsrc0, 0, 1);

  // -------- Phase 1a: LayerNorm (4 children x 4 batch rows per wave) + residual
  const float4 g0  = *(const float4*)(gamma + lane * 4);
  const float4 g1  = *(const float4*)(gamma + 128 + lane * 4);
  const float4 be0 = *(const float4*)(beta + lane * 4);
  const float4 be1 = *(const float4*)(beta + 128 + lane * 4);

  for (int bq = 0; bq < 4; ++bq) {
    const int b = wave * 4 + bq;
    float4 ra0 = {0.f, 0.f, 0.f, 0.f};
    float4 ra1 = {0.f, 0.f, 0.f, 0.f};
    #pragma unroll
    for (int c = 0; c < 4; ++c) {
      const float* row = x + (((size_t)b * N_ + n) * 4 + (size_t)c) * DIM_;
      const float4 v0 = *(const float4*)(row + lane * 4);
      const float4 v1 = *(const float4*)(row + 128 + lane * 4);
      ra0.x += v0.x; ra0.y += v0.y; ra0.z += v0.z; ra0.w += v0.w;
      ra1.x += v1.x; ra1.y += v1.y; ra1.z += v1.z; ra1.w += v1.w;
      float s  = v0.x + v0.y + v0.z + v0.w + v1.x + v1.y + v1.z + v1.w;
      float s2 = v0.x*v0.x + v0.y*v0.y + v0.z*v0.z + v0.w*v0.w
               + v1.x*v1.x + v1.y*v1.y + v1.z*v1.z + v1.w*v1.w;
      #pragma unroll
      for (int o = 16; o > 0; o >>= 1) {
        s  += __shfl_xor(s, o, 32);
        s2 += __shfl_xor(s2, o, 32);
      }
      const float mu   = s * (1.0f / 256.0f);
      const float var  = s2 * (1.0f / 256.0f) - mu * mu;
      const float rstd = rsqrtf(var + LN_EPS);
      const float t0 = (v0.x - mu) * rstd * g0.x + be0.x;
      const float t1 = (v0.y - mu) * rstd * g0.y + be0.y;
      const float t2 = (v0.z - mu) * rstd * g0.z + be0.z;
      const float t3 = (v0.w - mu) * rstd * g0.w + be0.w;
      const float t4 = (v1.x - mu) * rstd * g1.x + be1.x;
      const float t5 = (v1.y - mu) * rstd * g1.y + be1.y;
      const float t6 = (v1.z - mu) * rstd * g1.z + be1.z;
      const float t7 = (v1.w - mu) * rstd * g1.w + be1.w;
      unsigned short* xr = xn + (c * 32 + b) * 264;
      uint2 u01; u01.x = pk2(t0, t1); u01.y = pk2(t2, t3);
      uint2 u23; u23.x = pk2(t4, t5); u23.y = pk2(t6, t7);
      *(uint2*)(xr + lane * 4)       = u01;
      *(uint2*)(xr + 128 + lane * 4) = u23;
    }
    float* rr = res + b * 260;
    float4 w0; w0.x = ra0.x*0.25f; w0.y = ra0.y*0.25f; w0.z = ra0.z*0.25f; w0.w = ra0.w*0.25f;
    float4 w1; w1.x = ra1.x*0.25f; w1.y = ra1.y*0.25f; w1.z = ra1.z*0.25f; w1.w = ra1.w*0.25f;
    *(float4*)(rr + lane * 4)       = w0;
    *(float4*)(rr + 128 + lane * 4) = w1;
  }

  // -------- Phase 1b: stage factors to LDS as bf16
  const float* fsrc[4] = {fsrc0, fsrc1, fsrc2, fsrc3};
  #pragma unroll
  for (int c = 0; c < 4; ++c) {
    const float* src = fsrc[c];
    unsigned short* dst = fc + c * 8448;
    #pragma unroll
    for (int t = 0; t < 8; ++t) {
      const int e = (tid + t * 256) * 4;          // element index, [r][k] row-major
      const int r = e >> 8, k = e & 255;
      const float4 v = *(const float4*)(src + e);
      uint2 u; u.x = pk2(v.x, v.y); u.y = pk2(v.z, v.w);
      *(uint2*)(dst + r * 264 + k) = u;
    }
  }
  #pragma unroll
  for (int t = 0; t < 8; ++t) {                   // factor_out: transpose [r][o] -> [o][r]
    const int e = (tid + t * 256) * 4;
    const int r = e >> 8, o = e & 255;
    const float4 v = *(const float4*)(fsrco + e);
    fout[(o + 0) * 40 + r] = f2bf(v.x);
    fout[(o + 1) * 40 + r] = f2bf(v.y);
    fout[(o + 2) * 40 + r] = f2bf(v.z);
    fout[(o + 3) * 40 + r] = f2bf(v.w);
  }
  if (tid < 32) sc[tid] = scale[n * RANK_ + tid];
  __syncthreads();

  // -------- Phase 2: GEMM1  P_c[32,32] = XN_c[32,256] x F_c^T[256,32]
  {
    const int c     = wave >> 1;
    const int mtile = wave & 1;
    v8f acc0 = {0.f,0.f,0.f,0.f,0.f,0.f,0.f,0.f};
    v8f acc1 = {0.f,0.f,0.f,0.f,0.f,0.f,0.f,0.f};
    const unsigned short* fcc = fc + c * 8448;
    const unsigned short* ab  = xn + (c * 32 + mtile * 16 + l15) * 264 + (hiL ? 16 : 0);
    const unsigned short* bb0 = fcc + l15 * 264 + (hiL ? 16 : 0);
    const unsigned short* bb1 = fcc + (16 + l15) * 264 + (hiL ? 16 : 0);
    #pragma unroll
    for (int kb = 0; kb < 8; ++kb) {
      const v16bf a  = frag_a(ab  + kb * 32);
      const v16bf b0 = frag_b(bb0 + kb * 32);
      const v16bf b1 = frag_b(bb1 + kb * 32);
      acc0 = __builtin_amdgcn_wmma_f32_16x16x32_bf16(false, a, false, b0, (short)0, acc0, false, false);
      acc1 = __builtin_amdgcn_wmma_f32_16x16x32_bf16(false, a, false, b1, (short)0, acc1, false, false);
    }
    const int mb = mtile * 16 + (hiL ? 8 : 0);    // C/D layout: VGPR v -> row mb+v, col l15
    #pragma unroll
    for (int v = 0; v < 8; ++v) {
      pm[(c * 32 + mb + v) * 33 + l15]      = acc0[v];
      pm[(c * 32 + mb + v) * 33 + 16 + l15] = acc1[v];
    }
  }
  __syncthreads();

  // -------- Phase 3: merged[b,r] = scale[r] * prod_c P_c[b,r]  (bf16, A layout)
  #pragma unroll
  for (int t = 0; t < 4; ++t) {
    const int e = tid + t * 256;
    const int b = e >> 5, r = e & 31;
    const float m = sc[r]
                  * pm[(0 * 32 + b) * 33 + r]
                  * pm[(1 * 32 + b) * 33 + r]
                  * pm[(2 * 32 + b) * 33 + r]
                  * pm[(3 * 32 + b) * 33 + r];
    mg[b * 40 + r] = f2bf(m);
  }
  __syncthreads();

  // -------- Phase 4: GEMM2  OUT[32,256] = MERGED[32,32] x F_out[32,256]  (+ residual)
  {
    const int mtile = wave & 1;
    const int nb    = (wave >> 1) * 4;
    const v16bf a = frag_a(mg + (mtile * 16 + l15) * 40 + (hiL ? 16 : 0));
    const int mb  = mtile * 16 + (hiL ? 8 : 0);
    #pragma unroll
    for (int t = 0; t < 4; ++t) {
      const int o = (nb + t) * 16 + l15;
      const v16bf bf = frag_b(fout + o * 40 + (hiL ? 16 : 0));
      v8f z = {0.f,0.f,0.f,0.f,0.f,0.f,0.f,0.f};
      v8f d = __builtin_amdgcn_wmma_f32_16x16x32_bf16(false, a, false, bf, (short)0, z, false, false);
      #pragma unroll
      for (int v = 0; v < 8; ++v) {
        const int bi = mb + v;
        out[((size_t)bi * N_ + n) * DIM_ + o] = d[v] + res[bi * 260 + o];
      }
    }
  }
}

extern "C" void kernel_launch(void* const* d_in, const int* in_sizes, int n_in,
                              void* d_out, int out_size, void* d_ws, size_t ws_size,
                              hipStream_t stream) {
  (void)in_sizes; (void)n_in; (void)out_size; (void)d_ws; (void)ws_size;
  const float* x     = (const float*)d_in[0];
  const float* f_tl  = (const float*)d_in[1];
  const float* f_tr  = (const float*)d_in[2];
  const float* f_bl  = (const float*)d_in[3];
  const float* f_br  = (const float*)d_in[4];
  const float* f_out = (const float*)d_in[5];
  const float* scale = (const float*)d_in[6];
  const float* gamma = (const float*)d_in[7];
  const float* beta  = (const float*)d_in[8];
  float* out = (float*)d_out;
  cp_quad_rank_wmma<<<dim3(N_), dim3(256), LDS_BYTES, stream>>>(
      x, f_tl, f_tr, f_bl, f_br, f_out, scale, gamma, beta, out);
}
